// Qwen2MoeSparseMoeBlock_76879914598702
// MI455X (gfx1250) — matrix operand order repaired and checked by reading, not compiled, not run on hardware
//
#include <hip/hip_runtime.h>

typedef __attribute__((ext_vector_type(16))) _Float16 v16h;
typedef __attribute__((ext_vector_type(8)))  _Float16 v8h;
typedef __attribute__((ext_vector_type(8)))  float    v8f;
typedef __attribute__((ext_vector_type(4)))  float    v4f;

#define T_TOK 1024
#define H_DIM 2048
#define I_DIM 1408
#define SI_DIM 5632
#define NEXP 8

// ---------------------------------------------------------------- utilities

__device__ __forceinline__ v16h frag_a_from_lds(const _Float16* tile_row32, int lane) {
    // A 16x32 f16 fragment: lane holds row (lane&15); K chunks (lane>=16 ? 8 : 0)
    // in halves [0..7] and same +16 in halves [8..15]  (ISA 7.12.2, 16-bit A)
    const int r  = lane & 15;
    const int kb = (lane >> 4) << 3;               // 0 or 8
    const v8h lo = *(const v8h*)(tile_row32 + r * 32 + kb);
    const v8h hi = *(const v8h*)(tile_row32 + r * 32 + kb + 16);
    v16h f;
#pragma unroll
    for (int i = 0; i < 8; ++i) { f[i] = lo[i]; f[i + 8] = hi[i]; }
    return f;
}

__device__ __forceinline__ v16h frag_b_from_lds(const _Float16* tile_row32, int lane) {
    // B 32x16 f16 fragment: lane holds column (lane&15); K chunks (lane>=16 ? 8 : 0)
    // in halves [0..7] and same +16 in halves [8..15] (matches A layout)
    const int n  = lane & 15;
    const int kb = (lane >> 4) << 3;               // 0 or 8
    const v8h lo = *(const v8h*)(tile_row32 + n * 32 + kb);
    const v8h hi = *(const v8h*)(tile_row32 + n * 32 + kb + 16);
    v16h f;
#pragma unroll
    for (int i = 0; i < 8; ++i) { f[i] = lo[i]; f[i + 8] = hi[i]; }
    return f;
}

// ---------------------------------------------------------------- fp32 -> f16

__global__ __launch_bounds__(256) void cvt_hs_kernel(const float* __restrict__ in,
                                                     _Float16* __restrict__ out) {
    const size_t i = (size_t)(blockIdx.x * 256 + threadIdx.x) * 8;
    const v4f a = *(const v4f*)(in + i);
    const v4f b = *(const v4f*)(in + i + 4);
    v8h o;
#pragma unroll
    for (int j = 0; j < 4; ++j) { o[j] = (_Float16)a[j]; o[j + 4] = (_Float16)b[j]; }
    *(v8h*)(out + i) = o;
}

// fp32 [K,N] row-major  ->  f16 [N,K] row-major ("B^T" staging), batched over z
__global__ __launch_bounds__(256) void transpose_cvt_kernel(const float* __restrict__ in,
                                                            _Float16* __restrict__ out,
                                                            int K, int N) {
    __shared__ float tile[32][33];
    const float* inb  = in  + (size_t)blockIdx.z * K * N;
    _Float16*    outb = out + (size_t)blockIdx.z * K * N;
    const int k0 = blockIdx.x * 32, n0 = blockIdx.y * 32;
    const int tx = threadIdx.x & 31, ty = threadIdx.x >> 5;
#pragma unroll
    for (int r = ty; r < 32; r += 8) {
        const int k = k0 + r, n = n0 + tx;
        tile[r][tx] = (k < K && n < N) ? inb[(size_t)k * N + n] : 0.f;
    }
    __syncthreads();
#pragma unroll
    for (int r = ty; r < 32; r += 8) {
        const int n = n0 + r, k = k0 + tx;
        if (n < N && k < K) outb[(size_t)n * K + k] = (_Float16)tile[tx][r];
    }
}

// ---------------------------------------------------------------- router

__global__ __launch_bounds__(256) void router_kernel(const float* __restrict__ hs,
                                                     const float* __restrict__ gate_w,
                                                     const float* __restrict__ shared_gate_w,
                                                     float* __restrict__ logits_out,
                                                     float* __restrict__ shared_scale,
                                                     int* __restrict__ counts,
                                                     int* __restrict__ lists,
                                                     float* __restrict__ slot_w) {
    const int wave = threadIdx.x >> 5;
    const int lane = threadIdx.x & 31;
    const int t = blockIdx.x * 8 + wave;
    const float* x = hs + (size_t)t * H_DIM;

    float acc[NEXP];
#pragma unroll
    for (int e = 0; e < NEXP; ++e) acc[e] = 0.f;
    float accg = 0.f;
    for (int h = lane; h < H_DIM; h += 32) {
        const float xv = x[h];
        const float* gr = gate_w + (size_t)h * NEXP;
#pragma unroll
        for (int e = 0; e < NEXP; ++e) acc[e] += xv * gr[e];
        accg += xv * shared_gate_w[h];
    }
#pragma unroll
    for (int off = 16; off > 0; off >>= 1) {
#pragma unroll
        for (int e = 0; e < NEXP; ++e) acc[e] += __shfl_xor(acc[e], off, 32);
        accg += __shfl_xor(accg, off, 32);
    }
    if (lane == 0) {
        float mx = acc[0];
#pragma unroll
        for (int e = 1; e < NEXP; ++e) mx = fmaxf(mx, acc[e]);
        float p[NEXP], se = 0.f;
#pragma unroll
        for (int e = 0; e < NEXP; ++e) { p[e] = expf(acc[e] - mx); se += p[e]; }
        const float inv = 1.f / se;
        int b0 = 0;
#pragma unroll
        for (int e = 1; e < NEXP; ++e) if (p[e] > p[b0]) b0 = e;
        int b1 = -1;
#pragma unroll
        for (int e = 0; e < NEXP; ++e) {
            if (e == b0) continue;
            if (b1 < 0 || p[e] > p[b1]) b1 = e;
        }
#pragma unroll
        for (int e = 0; e < NEXP; ++e) logits_out[(size_t)t * NEXP + e] = acc[e];
        shared_scale[t] = 1.f / (1.f + expf(-accg));
        const int sel[2] = {b0, b1};
#pragma unroll
        for (int k = 0; k < 2; ++k) {
            const int e = sel[k];
            const int slot = atomicAdd(&counts[e], 1);
            lists[e * T_TOK + slot]  = t;
            slot_w[e * T_TOK + slot] = p[e] * inv;   // norm_topk_prob=False
        }
    }
}

// ---------------------------------------------------------------- FFN stage 1
// act[slot, n] = silu(A@Wg)[slot,n] * (A@Wu)[slot,n], f16 out.
// A rows gathered through `lists` (experts) or identity (shared expert).

__global__ __launch_bounds__(256) void ffn1_kernel(const _Float16* __restrict__ A,   // [T,K]
                                                   const _Float16* __restrict__ BgT, // [z,N,K]
                                                   const _Float16* __restrict__ BuT, // [z,N,K]
                                                   _Float16* __restrict__ Act,       // [z,T,N]
                                                   const int* __restrict__ lists,
                                                   const int* __restrict__ counts,
                                                   int K, int N) {
    __shared__ __align__(16) _Float16 a_sm[64 * 32];
    __shared__ __align__(16) _Float16 bg_sm[32 * 32];
    __shared__ __align__(16) _Float16 bu_sm[32 * 32];

    const int e   = blockIdx.z;
    const int cnt = counts ? counts[e] : T_TOK;
    const int m0  = blockIdx.y * 64;
    if (m0 >= cnt) return;
    const int n0 = blockIdx.x * 32;
    const int* list = lists ? lists + e * T_TOK : nullptr;
    const _Float16* Bg = BgT + (size_t)e * N * K;
    const _Float16* Bu = BuT + (size_t)e * N * K;
    _Float16* ActE = Act + (size_t)e * T_TOK * N;

    const int tid = threadIdx.x;
    // cooperative A load: 64 rows x 32 halves, 8 halves/thread
    const int  a_r    = tid >> 2;
    const int  a_chk  = (tid & 3) * 8;
    const int  a_slot = m0 + a_r;
    const bool a_ok   = a_slot < cnt;
    const int  a_row  = a_ok ? (list ? list[a_slot] : a_slot) : 0;
    const _Float16* a_gp = A + (size_t)a_row * K + a_chk;
    // cooperative B load: tid 0-127 -> Bg, 128-255 -> Bu
    const int  btid  = tid & 127;
    const int  b_r   = btid >> 2;
    const int  b_chk = (btid & 3) * 8;
    const _Float16* b_gp = ((tid < 128) ? Bg : Bu) + (size_t)(n0 + b_r) * K + b_chk;
    _Float16* b_sm = (tid < 128) ? bg_sm : bu_sm;

    const int lane = tid & 31, wv = tid >> 5;
    const int mw = wv >> 1, nw = wv & 1;
    const _Float16* a_fr  = a_sm  + (mw * 16) * 32;
    const _Float16* bg_fr = bg_sm + (nw * 16) * 32;
    const _Float16* bu_fr = bu_sm + (nw * 16) * 32;

    v8f cg = {}, cu = {};
    for (int k0 = 0; k0 < K; k0 += 32) {
        const v8h zz = {};
        const v8h av = a_ok ? *(const v8h*)(a_gp + k0) : zz;
        const v8h bv = *(const v8h*)(b_gp + k0);
        __syncthreads();
        *(v8h*)(a_sm + a_r * 32 + a_chk) = av;
        *(v8h*)(b_sm + b_r * 32 + b_chk) = bv;
        __syncthreads();
        const v16h af  = frag_a_from_lds(a_fr, lane);
        const v16h bgf = frag_b_from_lds(bg_fr, lane);
        const v16h buf = frag_b_from_lds(bu_fr, lane);
        cg = __builtin_amdgcn_wmma_f32_16x16x32_f16(false, af, false, bgf, (short)0, cg, false, false);
        cu = __builtin_amdgcn_wmma_f32_16x16x32_f16(false, af, false, buf, (short)0, cu, false, false);
    }

    const int nl = lane & 15;
    const int mb = (lane >> 4) << 3;
    const int col = n0 + nw * 16 + nl;
#pragma unroll
    for (int j = 0; j < 8; ++j) {
        const int slot = m0 + mw * 16 + mb + j;
        if (slot < cnt) {
            const float g = cg[j], u = cu[j];
            const float a = (g / (1.f + expf(-g))) * u;   // silu(g)*u
            ActE[(size_t)slot * N + col] = (_Float16)a;
        }
    }
}

// ---------------------------------------------------------------- FFN stage 2
// out[tok, n] += scale(slot) * (Act @ WdT^T)[slot, n]

__global__ __launch_bounds__(256) void ffn2_kernel(const _Float16* __restrict__ ActAll, // [z,T,K]
                                                   const _Float16* __restrict__ BdT,    // [z,N,K]
                                                   float* __restrict__ Out,             // [T,H]
                                                   const int* __restrict__ lists,
                                                   const int* __restrict__ counts,
                                                   const float* __restrict__ scale,
                                                   int K, int N, int scale_by_slot) {
    __shared__ __align__(16) _Float16 a_sm[64 * 32];
    __shared__ __align__(16) _Float16 b_sm[32 * 32];

    const int e   = blockIdx.z;
    const int cnt = counts ? counts[e] : T_TOK;
    const int m0  = blockIdx.y * 64;
    if (m0 >= cnt) return;
    const int n0 = blockIdx.x * 32;
    const int* list = lists ? lists + e * T_TOK : nullptr;
    const _Float16* Aa = ActAll + (size_t)e * T_TOK * K;
    const _Float16* Bd = BdT + (size_t)e * N * K;

    const int tid = threadIdx.x;
    const int  a_r    = tid >> 2;
    const int  a_chk  = (tid & 3) * 8;
    const int  a_slot = m0 + a_r;
    const bool a_ok   = a_slot < cnt;
    const _Float16* a_gp = Aa + (size_t)(a_ok ? a_slot : 0) * K + a_chk;
    const int  btid  = tid & 127;
    const int  b_r   = btid >> 2;
    const int  b_chk = (btid & 3) * 8;
    const _Float16* b_gp = Bd + (size_t)(n0 + b_r) * K + b_chk;

    const int lane = tid & 31, wv = tid >> 5;
    const int mw = wv >> 1, nw = wv & 1;
    const _Float16* a_fr = a_sm + (mw * 16) * 32;
    const _Float16* b_fr = b_sm + (nw * 16) * 32;

    v8f c = {};
    for (int k0 = 0; k0 < K; k0 += 32) {
        const v8h zz = {};
        const v8h av = a_ok ? *(const v8h*)(a_gp + k0) : zz;
        v8h bv = zz;
        if (tid < 128) bv = *(const v8h*)(b_gp + k0);
        __syncthreads();
        *(v8h*)(a_sm + a_r * 32 + a_chk) = av;
        if (tid < 128) *(v8h*)(b_sm + b_r * 32 + b_chk) = bv;
        __syncthreads();
        const v16h af = frag_a_from_lds(a_fr, lane);
        const v16h bf = frag_b_from_lds(b_fr, lane);
        c = __builtin_amdgcn_wmma_f32_16x16x32_f16(false, af, false, bf, (short)0, c, false, false);
    }

    const int nl = lane & 15;
    const int mb = (lane >> 4) << 3;
    const int col = n0 + nw * 16 + nl;
#pragma unroll
    for (int j = 0; j < 8; ++j) {
        const int slot = m0 + mw * 16 + mb + j;
        if (slot < cnt) {
            const int tok = list ? list[e * T_TOK + slot] : slot;
            const float s = scale[(scale_by_slot ? (e * T_TOK + slot) : tok)];
            atomicAdd(&Out[(size_t)tok * H_DIM + col], s * c[j]);
        }
    }
}

// ---------------------------------------------------------------- launch

static constexpr size_t alignup(size_t x) { return (x + 255) & ~(size_t)255; }

extern "C" void kernel_launch(void* const* d_in, const int* in_sizes, int n_in,
                              void* d_out, int out_size, void* d_ws, size_t ws_size,
                              hipStream_t stream) {
    (void)in_sizes; (void)n_in; (void)out_size; (void)ws_size;
    const float* hs            = (const float*)d_in[0];
    const float* gate_w        = (const float*)d_in[1];
    const float* w_gate        = (const float*)d_in[2];
    const float* w_up          = (const float*)d_in[3];
    const float* w_down        = (const float*)d_in[4];
    const float* sw_gate       = (const float*)d_in[5];
    const float* sw_up         = (const float*)d_in[6];
    const float* sw_down       = (const float*)d_in[7];
    const float* shared_gate_w = (const float*)d_in[8];

    float* out    = (float*)d_out;                       // [T,H]
    float* logits = out + (size_t)T_TOK * H_DIM;         // [T,E]

    char* ws = (char*)d_ws;
    const size_t o_hsf16  = 0;
    const size_t o_wgT    = o_hsf16 + alignup((size_t)T_TOK * H_DIM * 2);
    const size_t o_wuT    = o_wgT   + alignup((size_t)NEXP * I_DIM * H_DIM * 2);
    const size_t o_wdT    = o_wuT   + alignup((size_t)NEXP * I_DIM * H_DIM * 2);
    const size_t o_swgT   = o_wdT   + alignup((size_t)NEXP * I_DIM * H_DIM * 2);
    const size_t o_swuT   = o_swgT  + alignup((size_t)SI_DIM * H_DIM * 2);
    const size_t o_swdT   = o_swuT  + alignup((size_t)SI_DIM * H_DIM * 2);
    const size_t o_sscale = o_swdT  + alignup((size_t)H_DIM * SI_DIM * 2);
    const size_t o_counts = o_sscale + alignup((size_t)T_TOK * 4);
    const size_t o_lists  = o_counts + alignup((size_t)NEXP * 4);
    const size_t o_slotw  = o_lists  + alignup((size_t)NEXP * T_TOK * 4);
    const size_t o_acte   = o_slotw  + alignup((size_t)NEXP * T_TOK * 4);
    const size_t o_acts   = o_acte   + alignup((size_t)NEXP * T_TOK * I_DIM * 2);

    _Float16* hsf16  = (_Float16*)(ws + o_hsf16);
    _Float16* wgT    = (_Float16*)(ws + o_wgT);
    _Float16* wuT    = (_Float16*)(ws + o_wuT);
    _Float16* wdT    = (_Float16*)(ws + o_wdT);
    _Float16* swgT   = (_Float16*)(ws + o_swgT);
    _Float16* swuT   = (_Float16*)(ws + o_swuT);
    _Float16* swdT   = (_Float16*)(ws + o_swdT);
    float*    sscale = (float*)(ws + o_sscale);
    int*      counts = (int*)(ws + o_counts);
    int*      lists  = (int*)(ws + o_lists);
    float*    slotw  = (float*)(ws + o_slotw);
    _Float16* act_e  = (_Float16*)(ws + o_acte);
    _Float16* act_s  = (_Float16*)(ws + o_acts);

    hipMemsetAsync(out, 0, (size_t)T_TOK * H_DIM * sizeof(float), stream);
    hipMemsetAsync(counts, 0, NEXP * sizeof(int), stream);

    cvt_hs_kernel<<<(T_TOK * H_DIM) / (256 * 8), 256, 0, stream>>>(hs, hsf16);

    transpose_cvt_kernel<<<dim3(H_DIM / 32, I_DIM / 32, NEXP), 256, 0, stream>>>(w_gate, wgT, H_DIM, I_DIM);
    transpose_cvt_kernel<<<dim3(H_DIM / 32, I_DIM / 32, NEXP), 256, 0, stream>>>(w_up,   wuT, H_DIM, I_DIM);
    transpose_cvt_kernel<<<dim3(I_DIM / 32, H_DIM / 32, NEXP), 256, 0, stream>>>(w_down, wdT, I_DIM, H_DIM);
    transpose_cvt_kernel<<<dim3(H_DIM / 32, SI_DIM / 32, 1), 256, 0, stream>>>(sw_gate, swgT, H_DIM, SI_DIM);
    transpose_cvt_kernel<<<dim3(H_DIM / 32, SI_DIM / 32, 1), 256, 0, stream>>>(sw_up,   swuT, H_DIM, SI_DIM);
    transpose_cvt_kernel<<<dim3(SI_DIM / 32, H_DIM / 32, 1), 256, 0, stream>>>(sw_down, swdT, SI_DIM, H_DIM);

    router_kernel<<<T_TOK / 8, 256, 0, stream>>>(hs, gate_w, shared_gate_w,
                                                 logits, sscale, counts, lists, slotw);

    // sparse experts: gate/up -> silu*mul -> down, only over assigned tokens
    ffn1_kernel<<<dim3(I_DIM / 32, T_TOK / 64, NEXP), 256, 0, stream>>>(
        hsf16, wgT, wuT, act_e, lists, counts, H_DIM, I_DIM);
    ffn2_kernel<<<dim3(H_DIM / 32, T_TOK / 64, NEXP), 256, 0, stream>>>(
        act_e, wdT, out, lists, counts, slotw, I_DIM, H_DIM, 1);

    // shared expert (dense), sigmoid-gated
    ffn1_kernel<<<dim3(SI_DIM / 32, T_TOK / 64, 1), 256, 0, stream>>>(
        hsf16, swgT, swuT, act_s, nullptr, nullptr, H_DIM, SI_DIM);
    ffn2_kernel<<<dim3(H_DIM / 32, T_TOK / 64, 1), 256, 0, stream>>>(
        act_s, swdT, out, nullptr, nullptr, sscale, SI_DIM, H_DIM, 0);
}
